// GCN_10574209483250
// MI455X (gfx1250) — matrix operand order repaired and checked by reading, not compiled, not run on hardware
//
#include <hip/hip_runtime.h>
#include <stdint.h>

typedef __attribute__((ext_vector_type(16))) _Float16 v16h;
typedef __attribute__((ext_vector_type(8)))  _Float16 v8h;
typedef __attribute__((ext_vector_type(8)))  float    v8f;

#define FDIM 128          // input / hidden feature width
#define ODIM 64
#define GDIM 128
#define WT_STRIDE 136     // padded stride (f16 elems) -> 68 dwords, dodges LDS bank conflicts
#define WT_BYTES (FDIM * WT_STRIDE * 2)   // 34816 B, multiple of 16*128

// ---------------- elementwise helpers ----------------

__global__ void gcn_zero_f32(float* __restrict__ p, int n) {
    int i = blockIdx.x * blockDim.x + threadIdx.x;
    if (i < n) p[i] = 0.0f;
}

__global__ void gcn_cast_x(const float* __restrict__ x, _Float16* __restrict__ xh, int n) {
    int i = blockIdx.x * blockDim.x + threadIdx.x;
    if (i < n) xh[i] = (_Float16)x[i];
}

// W[k][c] f32 -> padded transposed f16 Wt[c*WT_STRIDE + k]
__global__ void gcn_cast_w(const float* __restrict__ W, _Float16* __restrict__ Wt) {
    int i = blockIdx.x * blockDim.x + threadIdx.x;
    if (i >= FDIM * FDIM) return;
    int k = i >> 7;
    int c = i & 127;
    Wt[c * WT_STRIDE + k] = (_Float16)W[i];
}

__global__ void gcn_degree(const int* __restrict__ dst, float* __restrict__ deg, int E) {
    int e = blockIdx.x * blockDim.x + threadIdx.x;
    if (e < E) atomicAdd(&deg[dst[e]], 1.0f);
}

__global__ void gcn_dinv(float* __restrict__ deg, int n) {
    int i = blockIdx.x * blockDim.x + threadIdx.x;
    if (i < n) deg[i] = rsqrtf(deg[i] + 1.0f);   // deg>0 always (self loop)
}

// ---------------- WMMA GEMM: C[N,128] = A_f16[N,128] * Wt_f16 ----------------
// block = 128 threads = 4 waves; each wave computes a 16x128 strip of C.
// W tile arrives pre-transposed/padded in global; staged to LDS via async copy.
__global__ __launch_bounds__(128) void gcn_gemm_wmma(
        const _Float16* __restrict__ A,     // [nrows,128] f16, row-major
        const _Float16* __restrict__ Wtg,   // [128][WT_STRIDE] f16, col-major padded
        float*          __restrict__ C,     // [nrows,128] f32
        int nrows) {
    __shared__ _Float16 Wt[FDIM * WT_STRIDE];

    const int tid = threadIdx.x;

    // Async DMA the 34 KB W tile into LDS (16 B per lane per op, 17 ops per thread).
    {
        unsigned ldsBase = (unsigned)(uintptr_t)(void*)Wt;
        const char* g = (const char*)Wtg;
        #pragma unroll
        for (int it = 0; it < WT_BYTES / (16 * 128); ++it) {
            int byteOff = (it * 128 + tid) * 16;
            unsigned l = ldsBase + (unsigned)byteOff;
            const char* gp = g + byteOff;
            asm volatile("global_load_async_to_lds_b128 %0, %1, off"
                         :: "v"(l), "v"(gp)
                         : "memory");
        }
#if __has_builtin(__builtin_amdgcn_s_wait_asynccnt)
        __builtin_amdgcn_s_wait_asynccnt(0);
#else
        asm volatile("s_wait_asynccnt 0x0" ::: "memory");
#endif
    }
    __syncthreads();

    const int wave = tid >> 5;
    const int lane = tid & 31;
    const int rowBase = blockIdx.x * 64 + wave * 16;
    if (rowBase >= nrows) return;             // uniform per wave; EXEC stays all-1s

    const int  laneLo = lane & 15;
    const bool hiHalf = lane >= 16;
    const _Float16* arow = A + (size_t)(rowBase + laneLo) * FDIM;

    v8f acc[8] = {};  // 8 column tiles of 16

    #pragma unroll
    for (int kc = 0; kc < 4; ++kc) {          // K chunks of 32
        // A fragment, contiguous K layout (matches B):
        //   lanes 0-15 : elems 0-15 = K kc*32+0..15
        //   lanes 16-31: elems 0-15 = K kc*32+16..31
        const int kb = kc * 32 + (hiHalf ? 16 : 0);
        v8h alo = *(const v8h*)(arow + kb);
        v8h ahi = *(const v8h*)(arow + kb + 8);
        v16h a;
        #pragma unroll
        for (int j = 0; j < 8; ++j) { a[j] = alo[j]; a[j + 8] = ahi[j]; }

        const int kbb = kc * 32 + (hiHalf ? 16 : 0);
        #pragma unroll
        for (int nt = 0; nt < 8; ++nt) {
            const int col = nt * 16 + laneLo;
            // B fragment: lane holds 16 consecutive K values of its column (from LDS).
            v16h b = *(const v16h*)(&Wt[col * WT_STRIDE + kbb]);
            acc[nt] = __builtin_amdgcn_wmma_f32_16x16x32_f16(
                false, a, false, b, (short)0, acc[nt], false, false);
        }
    }

    // C layout: lanes 0-15 hold rows 0-7 (VGPR r = row r), lanes 16-31 rows 8-15.
    const int rowOff = hiHalf ? 8 : 0;
    #pragma unroll
    for (int nt = 0; nt < 8; ++nt) {
        const int col = nt * 16 + laneLo;
        #pragma unroll
        for (int r = 0; r < 8; ++r)
            C[(size_t)(rowBase + rowOff + r) * FDIM + col] = acc[nt][r];
    }
}

// ---------------- edge aggregation: agg[dst] += dinv[src]*dinv[dst] * h[src] ----------------
// one wave per edge; each lane owns 4 consecutive features.
__global__ void gcn_aggregate(const int* __restrict__ src, const int* __restrict__ dst,
                              const float* __restrict__ dinv, const float* __restrict__ h,
                              float* __restrict__ agg, int E) {
    int g = blockIdx.x * blockDim.x + threadIdx.x;
    int e = g >> 5;
    int lane = g & 31;
    if (e >= E) return;
    if (lane == 0 && e + 2048 < E) {
        __builtin_prefetch(&src[e + 2048], 0, 0);   // global_prefetch_b8
        __builtin_prefetch(&dst[e + 2048], 0, 0);
    }
    int s = src[e];
    int d = dst[e];
    float nrm = dinv[s] * dinv[d];
    const float4 v = *(const float4*)(h + (size_t)s * FDIM + lane * 4);
    float* base = agg + (size_t)d * FDIM + lane * 4;
    atomicAdd(base + 0, nrm * v.x);
    atomicAdd(base + 1, nrm * v.y);
    atomicAdd(base + 2, nrm * v.z);
    atomicAdd(base + 3, nrm * v.w);
}

// act = relu(agg + dinv^2 * h + b); write f32 in place and f16 for next GEMM
__global__ void gcn_act(float* __restrict__ h, const float* __restrict__ agg,
                        const float* __restrict__ dinv, const float* __restrict__ b,
                        _Float16* __restrict__ xh, int n /* = N*128 */) {
    int i = blockIdx.x * blockDim.x + threadIdx.x;
    if (i >= n) return;
    int node = i >> 7;
    int f = i & 127;
    float di = dinv[node];
    float v = agg[i] + di * di * h[i] + b[f];
    v = fmaxf(v, 0.0f);
    h[i] = v;
    xh[i] = (_Float16)v;
}

// ---------------- pooling + final linear ----------------

__global__ void gcn_pool_cnt(const int* __restrict__ batch, float* __restrict__ cnt, int n) {
    int i = blockIdx.x * blockDim.x + threadIdx.x;
    if (i < n) atomicAdd(&cnt[batch[i]], 1.0f);
}

__global__ void gcn_pool_sum(const int* __restrict__ batch, const float* __restrict__ h,
                             float* __restrict__ sums, int n /* = N*128 */) {
    int i = blockIdx.x * blockDim.x + threadIdx.x;
    if (i >= n) return;
    int node = i >> 7;
    int f = i & 127;
    atomicAdd(&sums[(size_t)batch[node] * FDIM + f], h[i]);
}

__global__ void gcn_final(const float* __restrict__ sums, const float* __restrict__ cnt,
                          const float* __restrict__ linW, const float* __restrict__ linb,
                          float* __restrict__ out) {
    int i = blockIdx.x * blockDim.x + threadIdx.x;   // G*O threads
    if (i >= GDIM * ODIM) return;
    int gidx = i / ODIM;
    int o = i % ODIM;
    float c = fmaxf(cnt[gidx], 1.0f);
    float inv = 1.0f / c;
    float accv = 0.0f;
    #pragma unroll 8
    for (int k = 0; k < FDIM; ++k)
        accv += (sums[gidx * FDIM + k] * inv) * linW[k * ODIM + o];
    out[i] = accv + linb[o];
}

// ---------------- launcher ----------------

static inline size_t align256(size_t x) { return (x + 255) & ~(size_t)255; }

extern "C" void kernel_launch(void* const* d_in, const int* in_sizes, int n_in,
                              void* d_out, int out_size, void* d_ws, size_t ws_size,
                              hipStream_t stream) {
    const float* x    = (const float*)d_in[0];
    const int*   ei   = (const int*)d_in[1];
    const int*   batch= (const int*)d_in[2];
    const float* Ws[3] = {(const float*)d_in[3], (const float*)d_in[5], (const float*)d_in[7]};
    const float* bs[3] = {(const float*)d_in[4], (const float*)d_in[6], (const float*)d_in[8]};
    const float* linW = (const float*)d_in[9];
    const float* linb = (const float*)d_in[10];
    float* out = (float*)d_out;

    const int N = in_sizes[2];          // number of nodes
    const int E = in_sizes[1] / 2;      // number of edges
    const int NF = N * FDIM;
    const int* srcI = ei;
    const int* dstI = ei + E;

    // workspace carve-out
    char* ws = (char*)d_ws;
    size_t off = 0;
    _Float16* xh   = (_Float16*)(ws + off); off += align256((size_t)NF * sizeof(_Float16));
    float* h       = (float*)(ws + off);    off += align256((size_t)NF * sizeof(float));
    float* agg     = (float*)(ws + off);    off += align256((size_t)NF * sizeof(float));
    float* dinv    = (float*)(ws + off);    off += align256((size_t)N * sizeof(float));
    float* sums    = (float*)(ws + off);    off += align256((size_t)GDIM * FDIM * sizeof(float));
    float* cnt     = (float*)(ws + off);    off += align256((size_t)GDIM * sizeof(float));
    _Float16* wt16 = (_Float16*)(ws + off); off += align256((size_t)WT_BYTES);

    const int TB = 256;
    auto blk = [](long long n, int tb) { return (int)((n + tb - 1) / tb); };

    // activations (f16) for layer 0
    gcn_cast_x<<<blk(NF, TB), TB, 0, stream>>>(x, xh, NF);

    // symmetric-norm degree factors
    gcn_zero_f32<<<blk(N, TB), TB, 0, stream>>>(dinv, N);
    gcn_degree<<<blk(E, TB), TB, 0, stream>>>(dstI, dinv, E);
    gcn_dinv<<<blk(N, TB), TB, 0, stream>>>(dinv, N);

    const int gemmBlocks = (N + 63) / 64;
    const long long aggThreads = (long long)E * 32;

    for (int l = 0; l < 3; ++l) {
        gcn_cast_w<<<blk(FDIM * FDIM, TB), TB, 0, stream>>>(Ws[l], wt16);
        gcn_gemm_wmma<<<gemmBlocks, 128, 0, stream>>>(xh, wt16, h, N);
        gcn_zero_f32<<<blk(NF, TB), TB, 0, stream>>>(agg, NF);
        gcn_aggregate<<<blk(aggThreads, TB), TB, 0, stream>>>(srcI, dstI, dinv, h, agg, E);
        gcn_act<<<blk(NF, TB), TB, 0, stream>>>(h, agg, dinv, bs[l], xh, NF);
    }

    // global mean pool
    gcn_zero_f32<<<blk(GDIM * FDIM, TB), TB, 0, stream>>>(sums, GDIM * FDIM);
    gcn_zero_f32<<<blk(GDIM, TB), TB, 0, stream>>>(cnt, GDIM);
    gcn_pool_cnt<<<blk(N, TB), TB, 0, stream>>>(batch, cnt, N);
    gcn_pool_sum<<<blk(NF, TB), TB, 0, stream>>>(batch, h, sums, NF);

    // final linear -> [G, O]
    gcn_final<<<blk(GDIM * ODIM, TB), TB, 0, stream>>>(sums, cnt, linW, linb, out);
}
